// Create_recurrent_unit_72954314489889
// MI455X (gfx1250) — matrix operand order repaired and checked by reading, not compiled, not run on hardware
//
#include <hip/hip_runtime.h>
#include <hip/hip_bf16.h>

// ---------------------------------------------------------------------------
// LSTM cell on MI455X (gfx1250, wave32, WMMA).
//   gates = x@W + h@U + b  -> BF16 WMMA 16x16x32, f32 accumulate
//   memory-bound (~320MB @ 23.3TB/s => ~13.7us floor); weights bf16 in L2.
//   Block = 64 batch rows x 512 gate cols, 512 threads = 16 waves.
//   B fragments double-buffered; sched_barrier(0) between k-slices keeps
//   each slice's 8 independent WMMAs together while slice s+1's 16 B loads
//   issue in region s -> a full slice of latency cover per load.
// ---------------------------------------------------------------------------

typedef __attribute__((ext_vector_type(16))) __bf16 v16bf;
typedef __attribute__((ext_vector_type(8)))  float  v8f;

#define B_ROWS   131072
#define E_DIM    128
#define H_DIM    128
#define BM       64                  // batch rows per workgroup
#define NTHREADS 512                 // 16 waves
#define LDSROW   136                 // padded row stride (bf16 elems)
#define PACK_PER_MAT 65536           // 128*512 elements per packed matrix

__device__ __forceinline__ unsigned short f2bf(float f) {
  unsigned u = __float_as_uint(f);
  u += 0x7FFFu + ((u >> 16) & 1u);       // round-to-nearest-even
  return (unsigned short)(u >> 16);
}

struct U32x8 { uint4 lo, hi; };
__device__ __forceinline__ v16bf make_frag(uint4 lo, uint4 hi) {
  U32x8 t{lo, hi};
  return __builtin_bit_cast(v16bf, t);
}

__device__ __forceinline__ float fast_sigmoid(float x) {
  return __builtin_amdgcn_rcpf(1.0f + __expf(-x));
}
__device__ __forceinline__ float fast_tanh(float x) {
  float e = __expf(-2.0f * x);
  return (1.0f - e) * __builtin_amdgcn_rcpf(1.0f + e);
}

// ---------------------------------------------------------------------------
// Pack W=[Wi|Wf|Wo|Wc] and U=[Ui|Uf|Uo|Uc] (f32 row-major [128 x 512]) into
// bf16 WMMA-B fragments. Layout per (ktile,ntile) 32x16 block (1KB):
//   lane L (0..15):  col N=L,  K = kt*32 + 0..15  (16 contiguous bf16)
//   lane L+16:       col N=L,  K = kt*32 + 16..31
// Linear thread index == output address.
// ---------------------------------------------------------------------------
__global__ void pack_weights(const float* __restrict__ Wi, const float* __restrict__ Wf,
                             const float* __restrict__ Wo, const float* __restrict__ Wc,
                             const float* __restrict__ Ui, const float* __restrict__ Uf,
                             const float* __restrict__ Uo, const float* __restrict__ Uc,
                             unsigned short* __restrict__ wp) {
  int t = blockIdx.x * blockDim.x + threadIdx.x;
  if (t >= 2 * PACK_PER_MAT) return;
  int phase = t / PACK_PER_MAT;            // 0 = W (x path), 1 = U (h path)
  int idx   = t % PACK_PER_MAT;
  int kt    = idx / 16384;                 // k-tile (4, 32 K each)
  int rem   = idx % 16384;
  int tile  = rem / 512;                   // n-tile (32, 16 cols each)
  int r2    = rem % 512;
  int lane  = r2 / 16;
  int q     = r2 % 16;
  int col   = tile * 16 + (lane & 15);     // 0..511 concatenated gate column
  int k     = kt * 32 + ((lane >= 16) ? 16 : 0) + q;
  int gate  = col / H_DIM;
  int gcol  = col % H_DIM;
  const float* src;
  if (phase == 0) src = (gate == 0) ? Wi : (gate == 1) ? Wf : (gate == 2) ? Wo : Wc;
  else            src = (gate == 0) ? Ui : (gate == 1) ? Uf : (gate == 2) ? Uo : Uc;
  wp[t] = f2bf(src[k * H_DIM + gcol]);
}

// ---------------------------------------------------------------------------
// Main kernel. 16 waves: mt = wave&3 (16-row M-tile), nb = wave>>2 (32-col H
// group). Wave owns N-tiles {2nb,2nb+1} of each gate -> 8 accumulators,
// 64 WMMAs, full i,f,o,c set locally -> epilogue needs no exchange.
// K-slice s = phase*4 + kt; packed-B offset for slice s is s*16384.
// ---------------------------------------------------------------------------
__global__ void __launch_bounds__(NTHREADS, 1)
lstm_kernel(const float* __restrict__ x, const float* __restrict__ hidden,
            const unsigned short* __restrict__ wp,
            const float* __restrict__ bi_p, const float* __restrict__ bf_p,
            const float* __restrict__ bo_p, const float* __restrict__ bc_p,
            float* __restrict__ out) {
  __shared__ unsigned short sm[2 * BM * LDSROW];   // 34.8 KB
  unsigned short* sX = sm;
  unsigned short* sH = sm + BM * LDSROW;

  const int tid = threadIdx.x;
  const int m0  = blockIdx.x * BM;
  const float* hp = hidden;                                  // h_prev [B,H]
  const float* cp = hidden + (size_t)B_ROWS * H_DIM;         // c_prev [B,H]

  const int wave = tid >> 5;
  const int lane = tid & 31;
  const int mt   = wave & 3;             // which 16-row M-tile (0..3)
  const int nb   = wave >> 2;            // which 32-col H group (0..3)
  const int mrow = mt * 16 + (lane & 15);
  const int kSel = (lane >= 16) ? 16 : 0; // A-frag lane-half K offset (contiguous layout)
  const int lcol = lane & 15;
  const int radd = (lane >> 4) << 3;     // C-layout row add: +0 or +8

  // --- B fragment loader: slice s (0..7), local tile j (0..7) --------------
  auto loadB = [&](int s, int j) -> v16bf {
    int tile = 2 * nb + ((j >> 1) << 3) + (j & 1);   // 2nb,2nb+1,2nb+8,...
    const unsigned short* bp = wp + (size_t)s * 16384 + (size_t)tile * 512 + lane * 16;
    return make_frag(*(const uint4*)bp, *(const uint4*)(bp + 8));
  };

  // --- prefetch slice 0 B fragments + epilogue HBM reads (pre-barrier) ----
  v16bf bb[2][8];                         // double-buffered B fragments
  #pragma unroll
  for (int j = 0; j < 8; ++j) bb[0][j] = loadB(0, j);

  float cpre[2][8];                       // c_prev values for our C tiles
  float vb[2][4];                         // biases (i,f,o,c) x 2 col sub-tiles
  #pragma unroll
  for (int t = 0; t < 2; ++t) {
    int col = 32 * nb + 16 * t + lcol;
    vb[t][0] = bi_p[col]; vb[t][1] = bf_p[col];
    vb[t][2] = bo_p[col]; vb[t][3] = bc_p[col];
    #pragma unroll
    for (int r = 0; r < 8; ++r) {
      int row = m0 + mt * 16 + r + radd;
      cpre[t][r] = cp[(size_t)row * H_DIM + col];
    }
  }

  // --- stage x[m0:m0+64,:] and h_prev[m0:m0+64,:] into LDS as bf16 ---------
  #pragma unroll
  for (int it = 0; it < 4; ++it) {
    int fidx = it * NTHREADS + tid;
    int row  = fidx >> 5;                // 0..63
    int c4   = (fidx & 31);              // float4 column index
    float4 vx = ((const float4*)(x  + (size_t)(m0 + row) * E_DIM))[c4];
    float4 vh = ((const float4*)(hp + (size_t)(m0 + row) * H_DIM))[c4];
    uint2 px, ph;
    px.x = (unsigned)f2bf(vx.x) | ((unsigned)f2bf(vx.y) << 16);
    px.y = (unsigned)f2bf(vx.z) | ((unsigned)f2bf(vx.w) << 16);
    ph.x = (unsigned)f2bf(vh.x) | ((unsigned)f2bf(vh.y) << 16);
    ph.y = (unsigned)f2bf(vh.z) | ((unsigned)f2bf(vh.w) << 16);
    *(uint2*)&sX[row * LDSROW + c4 * 4] = px;
    *(uint2*)&sH[row * LDSROW + c4 * 4] = ph;
  }
  __syncthreads();

  // --- GEMM: 8 k-slices, double buffer, hard fence between slices ----------
  v8f acc[8];
  #pragma unroll
  for (int j = 0; j < 8; ++j) acc[j] = v8f{};

  #pragma unroll
  for (int s = 0; s < 8; ++s) {
    const unsigned short* sA = (s >= 4) ? sH : sX;
    const int kt  = s & 3;
    const int cur = s & 1;
    const int nxt = cur ^ 1;
    // A fragment (16x32 bf16): lanes 0-15 K=kt*32+{0..15},
    //                          lanes 16-31 K=kt*32+{16..31}
    const unsigned short* ap = &sA[mrow * LDSROW + kt * 32 + kSel];
    v16bf afrag = make_frag(*(const uint4*)ap, *(const uint4*)(ap + 8));

    if (s < 7) {
      #pragma unroll
      for (int j = 0; j < 8; ++j) bb[nxt][j] = loadB(s + 1, j);  // prefetch
    }
    #pragma unroll
    for (int j = 0; j < 8; ++j)
      acc[j] = __builtin_amdgcn_wmma_f32_16x16x32_bf16(
          false, afrag, false, bb[cur][j], (short)0, acc[j], false, false);

    // hard scheduling fence: keep this slice's 8 independent WMMAs together;
    // slice s+1's B loads were issued above and stay in this region.
    __builtin_amdgcn_sched_barrier(0);
  }

  // --- LSTM epilogue: acc[0..1]=i, acc[2..3]=f, acc[4..5]=o, acc[6..7]=c~ --
  float* hout = out;
  float* cout = out + (size_t)B_ROWS * H_DIM;
  #pragma unroll
  for (int t = 0; t < 2; ++t) {
    int col = 32 * nb + 16 * t + lcol;
    #pragma unroll
    for (int r = 0; r < 8; ++r) {
      int row = m0 + mt * 16 + r + radd;
      float iv = fast_sigmoid(acc[0 + t][r] + vb[t][0]);
      float fv = fast_sigmoid(acc[2 + t][r] + vb[t][1]);
      float ov = fast_sigmoid(acc[4 + t][r] + vb[t][2]);
      float gv = fast_tanh   (acc[6 + t][r] + vb[t][3]);
      float cn = fv * cpre[t][r] + iv * gv;
      float hn = ov * fast_tanh(cn);
      hout[(size_t)row * H_DIM + col] = hn;
      cout[(size_t)row * H_DIM + col] = cn;
    }
  }
}

extern "C" void kernel_launch(void* const* d_in, const int* in_sizes, int n_in,
                              void* d_out, int out_size, void* d_ws, size_t ws_size,
                              hipStream_t stream) {
  const float* x      = (const float*)d_in[0];
  const float* hidden = (const float*)d_in[1];
  const float* Wi = (const float*)d_in[2];
  const float* Ui = (const float*)d_in[3];
  const float* bi = (const float*)d_in[4];
  const float* Wf = (const float*)d_in[5];
  const float* Uf = (const float*)d_in[6];
  const float* bf = (const float*)d_in[7];
  const float* Wo = (const float*)d_in[8];
  const float* Uo = (const float*)d_in[9];
  const float* bo = (const float*)d_in[10];
  const float* Wc = (const float*)d_in[11];
  const float* Uc = (const float*)d_in[12];
  const float* bc = (const float*)d_in[13];
  float* out = (float*)d_out;
  unsigned short* wp = (unsigned short*)d_ws;   // 256 KB packed bf16 weights

  // 1) pack weights to bf16 WMMA-B layout (131072 elements)
  pack_weights<<<(2 * PACK_PER_MAT + 255) / 256, 256, 0, stream>>>(
      Wi, Wf, Wo, Wc, Ui, Uf, Uo, Uc, wp);

  // 2) fused gates GEMM + LSTM epilogue
  lstm_kernel<<<B_ROWS / BM, NTHREADS, 0, stream>>>(
      x, hidden, wp, bi, bf, bo, bc, out);
}
